// TreensformerBlock_BranchMLP_20950850470563
// MI455X (gfx1250) — matrix operand order repaired and checked by reading, not compiled, not run on hardware
//
#include <hip/hip_runtime.h>
#include <stdint.h>

// ---------------- problem constants ----------------
#define EMBD    768
#define NHEAD   12
#define HD      64
#define BATCH   4
#define T_NODES 2047
#define NLEV    11
#define NLEAF   1024
#define HIDDEN  3072
#define KCAT    (NLEV * EMBD)        // 8448
#define MTOK    (BATCH * T_NODES)    // 8188
#define MLEAF   (BATCH * NLEAF)      // 4096
#define QKVW    (3 * EMBD)           // 2304

// ---------------- CDNA5 WMMA types ----------------
typedef __bf16 bf16_t;
typedef bf16_t v16bf __attribute__((ext_vector_type(16)));
typedef float  v8f   __attribute__((ext_vector_type(8)));

union FragB16 {
  v16bf v;
  unsigned short u[16];
  uint4 q[2];
};

// async global->LDS path (gfx1250); hedge on builtin availability
#if defined(__has_builtin)
#if __has_builtin(__builtin_amdgcn_global_load_async_to_lds_b128) && \
    __has_builtin(__builtin_amdgcn_s_wait_asynccnt)
#define USE_ASYNC_LDS 1
#endif
#endif

#ifdef USE_ASYNC_LDS
typedef int v4i_a __attribute__((ext_vector_type(4)));
typedef __attribute__((address_space(1))) v4i_a gv4i_t;   // global v4i
typedef __attribute__((address_space(3))) v4i_a lv4i_t;   // LDS v4i
#define ASYNC_WAIT() __builtin_amdgcn_s_wait_asynccnt(0)
#else
#define ASYNC_WAIT() ((void)0)
#endif

__device__ __forceinline__ unsigned short f2bf(float f) {
  unsigned int u = __float_as_uint(f);
  u = (u + 0x7fffu + ((u >> 16) & 1u)) >> 16;   // round-to-nearest-even
  return (unsigned short)u;
}

__device__ __forceinline__ v8f wmma_bf16(const FragB16& a, const FragB16& b, v8f c) {
  return __builtin_amdgcn_wmma_f32_16x16x32_bf16(
      false, a.v, false, b.v, (short)0, c, false, false);
}

__device__ __forceinline__ bool tree_related(int i, int j) {
  int a = i < j ? i : j;
  int b = i < j ? j : i;
  while (b > a) b = (b - 1) >> 1;
  return b == a;
}

// ---------------- LayerNorm (f32 in -> bf16 out) ----------------
__global__ void ln_kernel(const float* __restrict__ src,
                          const float* __restrict__ w,
                          const float* __restrict__ b,
                          unsigned short* __restrict__ dst) {
  int row = blockIdx.x;
  const float* s = src + (size_t)row * EMBD;
  __shared__ float red[256];
  float acc = 0.f;
  for (int i = threadIdx.x; i < EMBD; i += 256) acc += s[i];
  red[threadIdx.x] = acc; __syncthreads();
  for (int off = 128; off > 0; off >>= 1) {
    if (threadIdx.x < off) red[threadIdx.x] += red[threadIdx.x + off];
    __syncthreads();
  }
  float mean = red[0] * (1.f / EMBD);
  __syncthreads();
  float var = 0.f;
  for (int i = threadIdx.x; i < EMBD; i += 256) { float d = s[i] - mean; var += d * d; }
  red[threadIdx.x] = var; __syncthreads();
  for (int off = 128; off > 0; off >>= 1) {
    if (threadIdx.x < off) red[threadIdx.x] += red[threadIdx.x + off];
    __syncthreads();
  }
  float rstd = rsqrtf(red[0] * (1.f / EMBD) + 1e-5f);
  unsigned short* d = dst + (size_t)row * EMBD;
  for (int i = threadIdx.x; i < EMBD; i += 256)
    d[i] = f2bf((s[i] - mean) * rstd * w[i] + b[i]);
}

// ---------------- pack f32 weight (K x N) into B-fragment order, bf16 ----
// value(lane,e) = W[kb*32 + (lane>>4)*8 + e + (e&8)][nt*16 + (lane&15)]
__global__ void pack_w_kernel(const float* __restrict__ W,
                              unsigned short* __restrict__ out, int N) {
  int Nt = N >> 4;
  int tile = blockIdx.x;
  int kb = tile / Nt, nt = tile - kb * Nt;
  int lane = threadIdx.x & 31;
  int hh = lane >> 4, l15 = lane & 15;
  const float* src = W + (size_t)(kb * 32 + hh * 8) * N + nt * 16 + l15;
  alignas(16) unsigned short tmp[16];
#pragma unroll
  for (int e = 0; e < 16; e++) tmp[e] = f2bf(src[(size_t)(e + (e & 8)) * N]);
  unsigned short* o = out + ((size_t)tile * 32 + lane) * 16;
  *(uint4*)(o)     = *(const uint4*)(tmp);
  *(uint4*)(o + 8) = *(const uint4*)(tmp + 8);
}

// ---------------- cooperative GEMM core ---------------------------------
// Block = 128 threads (4 waves). Block tile = 64 rows x 128 cols.
// Packed-B panel for one 32-deep K step (8 n-tiles = 4096 halfs = 8KB,
// contiguous) is staged into LDS (double buffered) via async loads and
// shared by all 4 waves. Each wave: 1 A-frag -> 8 WMMAs per K step.

__device__ __forceinline__ void stage_b(const unsigned short* __restrict__ gsrc,
                                        unsigned short* ldst, int tid) {
#pragma unroll
  for (int t = 0; t < 4; t++) {
    int j = tid + t * 128;                       // 512 x 16B chunks
#ifdef USE_ASYNC_LDS
    __builtin_amdgcn_global_load_async_to_lds_b128(
        (gv4i_t*)(gsrc + (size_t)j * 8), (lv4i_t*)(ldst + (size_t)j * 8), 0, 0);
#else
    uint4 v = *(const uint4*)(gsrc + (size_t)j * 8);
    *(uint4*)(ldst + (size_t)j * 8) = v;
#endif
  }
}

struct ALoadRow {
  const unsigned short* base;
  int K;
  __device__ __forceinline__ void operator()(FragB16& a, int row, int kb, int hh) const {
    const unsigned short* ap = base + (size_t)row * K + kb * 32 + hh * 8;
    a.q[0] = *(const uint4*)(ap);
    a.q[1] = *(const uint4*)(ap + 16);
  }
};

// branch-MLP path gather: A[row=(b,leaf)][k=(level n, dim d)] = h2[b, node(n,leaf), d]
struct ALoadGather {
  const unsigned short* h2;
  __device__ __forceinline__ void operator()(FragB16& a, int row, int kb, int hh) const {
    int k0 = kb * 32;
    int n = k0 / EMBD;                           // 768 % 32 == 0: no straddle
    int dbase = k0 - n * EMBD + hh * 8;
    int bb = row >> 10, leaf = row & 1023;
    int node = (1 << n) - 1 + (leaf >> (10 - n));
    const unsigned short* ap = h2 + ((size_t)(bb * T_NODES + node)) * EMBD + dbase;
    a.q[0] = *(const uint4*)(ap);
    a.q[1] = *(const uint4*)(ap + 16);
  }
};

template <typename ALoad>
__device__ __forceinline__ void coop_gemm(const ALoad& aload, int rowc,
                                          const unsigned short* __restrict__ Wp,
                                          int Ntiles, int nt0, int Kb,
                                          unsigned short* lds, v8f acc[8]) {
  int tid  = threadIdx.x;
  int lane = tid & 31, hh = lane >> 4;
  stage_b(Wp + ((size_t)nt0) * 512, lds, tid);   // kb = 0 -> buffer 0
  for (int kb = 0; kb < Kb; ++kb) {
    ASYNC_WAIT();
    __syncthreads();                             // current buffer ready
    unsigned short* cur = lds + (size_t)(kb & 1) * 4096;
    if (kb + 1 < Kb)
      stage_b(Wp + ((size_t)(kb + 1) * Ntiles + nt0) * 512,
              lds + (size_t)((kb + 1) & 1) * 4096, tid);
    FragB16 a;
    aload(a, rowc, kb, hh);
#pragma unroll
    for (int t = 0; t < 8; t++) {
      FragB16 bf;
      const unsigned short* p = cur + t * 512 + lane * 16;
      bf.q[0] = *(const uint4*)(p);
      bf.q[1] = *(const uint4*)(p + 8);
      acc[t] = wmma_bf16(a, bf, acc[t]);
    }
    __syncthreads();                             // done reading before restage
  }
}

// ---------------- QKV GEMM: (8188x768)x(768x2304) + bias -> bf16 --------
__global__ void __launch_bounds__(128)
gemm_qkv_kernel(const unsigned short* __restrict__ hbf,
                const unsigned short* __restrict__ Wp,
                const float* __restrict__ bias,
                unsigned short* __restrict__ out) {
  __shared__ unsigned short Bst[8192];
  int tid = threadIdx.x, lane = tid & 31, wv = tid >> 5;
  int l15 = lane & 15, hh = lane >> 4;
  int row = blockIdx.x * 64 + wv * 16 + l15;
  int rowc = row < MTOK ? row : MTOK - 1;
  int nt0 = blockIdx.y * 8;
  v8f acc[8] = {};
  ALoadRow al{hbf, EMBD};
  coop_gemm(al, rowc, Wp, QKVW / 16, nt0, EMBD / 32, Bst, acc);
#pragma unroll
  for (int t = 0; t < 8; t++) {
    int col = (nt0 + t) * 16 + l15;
    float bv = bias[col];
#pragma unroll
    for (int r = 0; r < 8; r++) {
      int m = blockIdx.x * 64 + wv * 16 + r + 8 * hh;
      if (m < MTOK) out[(size_t)m * QKVW + col] = f2bf(acc[t][r] + bv);
    }
  }
}

// ---------------- proj GEMM + residual: d_out = x + att@Wp + b ----------
__global__ void __launch_bounds__(128)
gemm_proj_kernel(const unsigned short* __restrict__ att,
                 const unsigned short* __restrict__ Wp,
                 const float* __restrict__ bias,
                 const float* __restrict__ xres,
                 float* __restrict__ out) {
  __shared__ unsigned short Bst[8192];
  int tid = threadIdx.x, lane = tid & 31, wv = tid >> 5;
  int l15 = lane & 15, hh = lane >> 4;
  int row = blockIdx.x * 64 + wv * 16 + l15;
  int rowc = row < MTOK ? row : MTOK - 1;
  int nt0 = blockIdx.y * 8;
  v8f acc[8] = {};
  ALoadRow al{att, EMBD};
  coop_gemm(al, rowc, Wp, EMBD / 16, nt0, EMBD / 32, Bst, acc);
#pragma unroll
  for (int t = 0; t < 8; t++) {
    int col = (nt0 + t) * 16 + l15;
    float bv = bias[col];
#pragma unroll
    for (int r = 0; r < 8; r++) {
      int m = blockIdx.x * 64 + wv * 16 + r + 8 * hh;
      if (m < MTOK) {
        size_t idx = (size_t)m * EMBD + col;
        out[idx] = xres[idx] + acc[t][r] + bv;
      }
    }
  }
}

// ---------------- branch-MLP hidden GEMM (fused gather) + GELU ----------
__global__ void __launch_bounds__(128)
gemm_hidden_kernel(const unsigned short* __restrict__ h2bf,
                   const unsigned short* __restrict__ Apk,
                   unsigned short* __restrict__ hid) {
  __shared__ unsigned short Bst[8192];
  int tid = threadIdx.x, lane = tid & 31, wv = tid >> 5;
  int l15 = lane & 15, hh = lane >> 4;
  int row = blockIdx.x * 64 + wv * 16 + l15;     // < 4096 always
  int nt0 = blockIdx.y * 8;
  v8f acc[8] = {};
  ALoadGather al{h2bf};
  coop_gemm(al, row, Apk, HIDDEN / 16, nt0, KCAT / 32, Bst, acc);
#pragma unroll
  for (int t = 0; t < 8; t++) {
    int col = (nt0 + t) * 16 + l15;
#pragma unroll
    for (int r = 0; r < 8; r++) {
      int m = blockIdx.x * 64 + wv * 16 + r + 8 * hh;
      float v = acc[t][r];
      float g = 0.5f * v * (1.f + tanhf(0.7978845608f * (v + 0.044715f * v * v * v)));
      hid[(size_t)m * HIDDEN + col] = f2bf(g);
    }
  }
}

// ---------------- output GEMM, scatter-add into leaf rows ---------------
__global__ void __launch_bounds__(128)
gemm_out_kernel(const unsigned short* __restrict__ hid,
                const unsigned short* __restrict__ Bpk,
                float* __restrict__ out) {
  __shared__ unsigned short Bst[8192];
  int tid = threadIdx.x, lane = tid & 31, wv = tid >> 5;
  int l15 = lane & 15, hh = lane >> 4;
  int row = blockIdx.x * 64 + wv * 16 + l15;     // < 4096 always
  int nt0 = blockIdx.y * 8;
  v8f acc[8] = {};
  ALoadRow al{hid, HIDDEN};
  coop_gemm(al, row, Bpk, EMBD / 16, nt0, HIDDEN / 32, Bst, acc);
#pragma unroll
  for (int t = 0; t < 8; t++) {
    int col = (nt0 + t) * 16 + l15;
#pragma unroll
    for (int r = 0; r < 8; r++) {
      int m = blockIdx.x * 64 + wv * 16 + r + 8 * hh;
      int b = m >> 10, leaf = m & 1023;
      size_t idx = ((size_t)(b * T_NODES + (NLEAF - 1) + leaf)) * EMBD + col;
      out[idx] += acc[t][r];                     // each (m,col) unique
    }
  }
}

// ---------------- flash tree attention, 16 queries per wave --------------
__global__ void attn_kernel(const unsigned short* __restrict__ qkv,
                            unsigned short* __restrict__ out) {
  int lane = threadIdx.x & 31, l15 = lane & 15, hh = lane >> 4;
  int qt = blockIdx.x, h = blockIdx.y, b = blockIdx.z;
  int qbase = qt * 16;
  __shared__ unsigned short Plds[16][40];

  int qrow = qbase + l15;
  int qc = qrow < T_NODES ? qrow : T_NODES - 1;
  const unsigned short* qp = qkv + ((size_t)(b * T_NODES + qc)) * QKVW + h * HD;
  FragB16 qf[2];
#pragma unroll
  for (int c = 0; c < 2; c++) {
    const unsigned short* p = qp + c * 32 + hh * 16;
    qf[c].q[0] = *(const uint4*)(p);
    qf[c].q[1] = *(const uint4*)(p + 8);
  }

  v8f O[4] = {};
  float mrun[8], lrun[8], alpha[8];
#pragma unroll
  for (int r = 0; r < 8; r++) { mrun[r] = -3.0e38f; lrun[r] = 0.f; }

  for (int kb = 0; kb < 64; ++kb) {
    int k0 = kb * 32;
    v8f S[2] = {};
#pragma unroll
    for (int nt = 0; nt < 2; ++nt) {
      int key = k0 + nt * 16 + l15;
      int kc = key < T_NODES ? key : T_NODES - 1;
      const unsigned short* kp =
          qkv + ((size_t)(b * T_NODES + kc)) * QKVW + EMBD + h * HD + hh * 16;
#pragma unroll
      for (int c = 0; c < 2; c++) {
        FragB16 bf;
        const unsigned short* p = kp + c * 32;
        bf.q[0] = *(const uint4*)(p);
        bf.q[1] = *(const uint4*)(p + 8);
        S[nt] = wmma_bf16(qf[c], bf, S[nt]);
      }
    }
#pragma unroll
    for (int r = 0; r < 8; r++) {
      int qg = qbase + r + 8 * hh;
      int key0 = k0 + l15, key1 = k0 + 16 + l15;
      bool qok = qg < T_NODES;
      float s0 = (qok && key0 < T_NODES && tree_related(qg, key0)) ? S[0][r] * 0.125f : -1.0e30f;
      float s1 = (qok && key1 < T_NODES && tree_related(qg, key1)) ? S[1][r] * 0.125f : -1.0e30f;
      float mb = fmaxf(s0, s1);
#pragma unroll
      for (int d = 1; d < 16; d <<= 1) mb = fmaxf(mb, __shfl_xor(mb, d, 32));
      float mnew = fmaxf(mrun[r], mb);
      float p0 = __expf(s0 - mnew), p1 = __expf(s1 - mnew);
      float rs = p0 + p1;
#pragma unroll
      for (int d = 1; d < 16; d <<= 1) rs += __shfl_xor(rs, d, 32);
      alpha[r] = __expf(mrun[r] - mnew);
      lrun[r] = lrun[r] * alpha[r] + rs;
      mrun[r] = mnew;
      Plds[r + 8 * hh][l15]      = f2bf(p0);
      Plds[r + 8 * hh][16 + l15] = f2bf(p1);
    }
    __syncthreads();
#pragma unroll
    for (int t = 0; t < 4; t++)
#pragma unroll
      for (int r = 0; r < 8; r++) O[t][r] *= alpha[r];

    FragB16 pf;
    pf.q[0] = *(const uint4*)(&Plds[l15][hh * 16]);
    pf.q[1] = *(const uint4*)(&Plds[l15][hh * 16 + 8]);
#pragma unroll
    for (int t = 0; t < 4; t++) {
      FragB16 vf;
#pragma unroll
      for (int e = 0; e < 16; e++) {
        int key = k0 + hh * 16 + e;
        int kc = key < T_NODES ? key : T_NODES - 1;
        unsigned short val =
            qkv[((size_t)(b * T_NODES + kc)) * QKVW + 2 * EMBD + h * HD + t * 16 + l15];
        vf.u[e] = key < T_NODES ? val : (unsigned short)0;
      }
      O[t] = wmma_bf16(pf, vf, O[t]);
    }
    __syncthreads();
  }
#pragma unroll
  for (int t = 0; t < 4; t++)
#pragma unroll
    for (int r = 0; r < 8; r++) {
      int row = qbase + r + 8 * hh;
      if (row < T_NODES)
        out[((size_t)(b * T_NODES + row)) * EMBD + h * HD + t * 16 + l15] =
            f2bf(O[t][r] / lrun[r]);
    }
}

// ---------------- launcher ----------------
extern "C" void kernel_launch(void* const* d_in, const int* in_sizes, int n_in,
                              void* d_out, int out_size, void* d_ws, size_t ws_size,
                              hipStream_t stream) {
  (void)in_sizes; (void)n_in; (void)out_size; (void)ws_size;
  const float* x      = (const float*)d_in[0];
  const float* ln1_w  = (const float*)d_in[1];
  const float* ln1_b  = (const float*)d_in[2];
  const float* w_qkv  = (const float*)d_in[3];
  const float* b_qkv  = (const float*)d_in[4];
  const float* w_proj = (const float*)d_in[5];
  const float* b_proj = (const float*)d_in[6];
  const float* ln2_w  = (const float*)d_in[7];
  const float* ln2_b  = (const float*)d_in[8];
  const float* A      = (const float*)d_in[9];
  const float* B_out  = (const float*)d_in[10];
  float* out = (float*)d_out;

  char* ws = (char*)d_ws;
  size_t off = 0;
  auto carve = [&](size_t bytes) -> char* {
    char* p = ws + off;
    off += (bytes + 255) & ~(size_t)255;
    return p;
  };
  unsigned short* pk_qkv  = (unsigned short*)carve((size_t)EMBD * QKVW * 2);
  unsigned short* pk_proj = (unsigned short*)carve((size_t)EMBD * EMBD * 2);
  unsigned short* pk_A    = (unsigned short*)carve((size_t)KCAT * HIDDEN * 2);
  unsigned short* pk_B    = (unsigned short*)carve((size_t)HIDDEN * EMBD * 2);
  unsigned short* hbf     = (unsigned short*)carve((size_t)MTOK * EMBD * 2);
  unsigned short* qkvbf   = (unsigned short*)carve((size_t)MTOK * QKVW * 2);
  unsigned short* attbf   = (unsigned short*)carve((size_t)MTOK * EMBD * 2);
  unsigned short* h2bf    = (unsigned short*)carve((size_t)MTOK * EMBD * 2);
  unsigned short* hidbf   = (unsigned short*)carve((size_t)MLEAF * HIDDEN * 2);

  pack_w_kernel<<<(EMBD / 32) * (QKVW / 16), 32, 0, stream>>>(w_qkv, pk_qkv, QKVW);
  pack_w_kernel<<<(EMBD / 32) * (EMBD / 16), 32, 0, stream>>>(w_proj, pk_proj, EMBD);
  pack_w_kernel<<<(KCAT / 32) * (HIDDEN / 16), 32, 0, stream>>>(A, pk_A, HIDDEN);
  pack_w_kernel<<<(HIDDEN / 32) * (EMBD / 16), 32, 0, stream>>>(B_out, pk_B, EMBD);

  ln_kernel<<<MTOK, 256, 0, stream>>>(x, ln1_w, ln1_b, hbf);
  gemm_qkv_kernel<<<dim3(128, QKVW / 128), 128, 0, stream>>>(hbf, pk_qkv, b_qkv, qkvbf);
  attn_kernel<<<dim3(128, NHEAD, BATCH), 32, 0, stream>>>(qkvbf, attbf);
  gemm_proj_kernel<<<dim3(128, EMBD / 128), 128, 0, stream>>>(attbf, pk_proj, b_proj, x, out);
  ln_kernel<<<MTOK, 256, 0, stream>>>(out, ln2_w, ln2_b, h2bf);
  gemm_hidden_kernel<<<dim3(MLEAF / 64, HIDDEN / 128), 128, 0, stream>>>(h2bf, pk_A, hidbf);
  gemm_out_kernel<<<dim3(MLEAF / 64, EMBD / 128), 128, 0, stream>>>(hidbf, pk_B, out);
}
